// Network_10058813407523
// MI455X (gfx1250) — matrix operand order repaired and checked by reading, not compiled, not run on hardware
//
#include <hip/hip_runtime.h>
#include <hip/hip_bf16.h>

typedef __attribute__((ext_vector_type(16))) __bf16 v16bf;
typedef __attribute__((ext_vector_type(2)))  __bf16 v2bf;
typedef __attribute__((ext_vector_type(8)))  float  v8f;
typedef __attribute__((ext_vector_type(4)))  int    v4i;

#define AS_GLOBAL __attribute__((address_space(1)))
#define AS_LOCAL  __attribute__((address_space(3)))

#if __has_builtin(__builtin_amdgcn_global_load_async_to_lds_b128)
#define HAVE_ASYNC_LDS 1
#else
#define HAVE_ASYNC_LDS 0
#endif

#if __has_builtin(__builtin_amdgcn_s_wait_asynccnt)
#define ASYNC_WAIT() __builtin_amdgcn_s_wait_asynccnt(0)
#else
#define ASYNC_WAIT() asm volatile("s_wait_asynccnt 0x0" ::: "memory")
#endif

// fp32 -> packed bf16 pair (RNE). Use the dedicated v_cvt_pk_bf16_f32 builtin when
// available so operand pairing is exact and no v_mov_b16 shuffles are needed.
__device__ __forceinline__ unsigned f2bf_pack(float a, float b) {
#if __has_builtin(__builtin_amdgcn_cvt_pk_bf16_f32)
    union { v2bf v; unsigned u; } p;
    p.v = __builtin_amdgcn_cvt_pk_bf16_f32(a, b);
    return p.u;
#else
    union { __bf16 h[2]; unsigned u; } p;
    p.h[0] = (__bf16)a;
    p.h[1] = (__bf16)b;
    return p.u;
#endif
}

__device__ __forceinline__ float bf2f(unsigned short h) {
    return __uint_as_float(((unsigned)h) << 16);
}

// C[M,N] = relu(A[M,K] @ W[N,K]^T + bias), output stored bf16.
// A fp32 (A_IS_BF16=false) or bf16 (true); W/bias fp32.
// Block: 256 threads = 8 waves; 128x128 tile; wave = 32x64 = 2x4 WMMA tiles.
template <bool A_IS_BF16>
__global__ __launch_bounds__(256)
void gemm_relu_bf16(const void* __restrict__ Araw, const float* __restrict__ W,
                    const float* __restrict__ bias, unsigned short* __restrict__ Out,
                    int M, int N, int K) {
    __shared__ __align__(16) unsigned short sA[128 * 72];  // 18 KB (pad 72)
    __shared__ __align__(16) unsigned short sB[128 * 72];  // 18 KB

    const int tid  = threadIdx.x;
    const int lane = tid & 31;
    const int wave = tid >> 5;
    const int wm = (wave & 3) * 32;   // wave row offset in block tile (4 x 32 = 128)
    const int wn = (wave >> 2) * 64;  // wave col offset in block tile (2 x 64 = 128)
    const int h  = lane >> 4;         // half-wave
    const int ln = lane & 15;

    const int rowBase = blockIdx.y * 128;
    const int colBase = blockIdx.x * 128;

    // Per-thread staging coords: tile is 128 rows x 64 K-elements = 1024 groups of 8.
    int rc[4], ldsOff[4];
#pragma unroll
    for (int q = 0; q < 4; ++q) {
        int g = tid + q * 256;
        int r = g >> 3, c = (g & 7) * 8;
        rc[q]     = r * K + c;    // loop-invariant: no 64-bit mul in the k0 loop
        ldsOff[q] = r * 72 + c;
    }

    const float*          Af = (const float*)Araw + (size_t)rowBase * K;
    const unsigned short* Ab = (const unsigned short*)Araw + (size_t)rowBase * K;
    const float*          Bf = W + (size_t)colBase * K;

    v8f acc[2][4];
#pragma unroll
    for (int i = 0; i < 2; ++i)
#pragma unroll
        for (int j = 0; j < 4; ++j)
            acc[i][j] = (v8f){0.f, 0.f, 0.f, 0.f, 0.f, 0.f, 0.f, 0.f};

    for (int k0 = 0; k0 < K; k0 += 64) {
        // ---- stage A tile 128x64 ----
        if (A_IS_BF16) {
#pragma unroll
            for (int q = 0; q < 4; ++q) {
                const unsigned short* gp = Ab + rc[q] + k0;
#if HAVE_ASYNC_LDS
                __builtin_amdgcn_global_load_async_to_lds_b128(
                    (AS_GLOBAL v4i*)(void*)gp,
                    (AS_LOCAL  v4i*)(void*)&sA[ldsOff[q]], 0, 0);
#else
                *(uint4*)(&sA[ldsOff[q]]) = *(const uint4*)gp;
#endif
            }
        } else {
#pragma unroll
            for (int q = 0; q < 4; ++q) {
                const float* gp = Af + rc[q] + k0;
                float4 v0 = *(const float4*)gp;
                float4 v1 = *(const float4*)(gp + 4);
                uint4 p;
                p.x = f2bf_pack(v0.x, v0.y);
                p.y = f2bf_pack(v0.z, v0.w);
                p.z = f2bf_pack(v1.x, v1.y);
                p.w = f2bf_pack(v1.z, v1.w);
                *(uint4*)(&sA[ldsOff[q]]) = p;
            }
        }
        // ---- stage B tile 128x64 (weights fp32 -> bf16) ----
#pragma unroll
        for (int q = 0; q < 4; ++q) {
            const float* gp = Bf + rc[q] + k0;
            float4 v0 = *(const float4*)gp;
            float4 v1 = *(const float4*)(gp + 4);
            uint4 p;
            p.x = f2bf_pack(v0.x, v0.y);
            p.y = f2bf_pack(v0.z, v0.w);
            p.z = f2bf_pack(v1.x, v1.y);
            p.w = f2bf_pack(v1.z, v1.w);
            *(uint4*)(&sB[ldsOff[q]]) = p;
        }
#if HAVE_ASYNC_LDS
        if (A_IS_BF16) ASYNC_WAIT();
#endif
        __syncthreads();

        // hint next K-panel (gfx1250 global_prefetch)
        if (k0 + 64 < K) {
            const void* pf = A_IS_BF16 ? (const void*)(Ab + rc[0] + k0 + 64)
                                       : (const void*)(Af + rc[0] + k0 + 64);
            __builtin_prefetch(pf, 0, 1);
            __builtin_prefetch((const void*)(Bf + rc[0] + k0 + 64), 0, 1);
        }

#pragma unroll
        for (int kk = 0; kk < 64; kk += 32) {
            // A frag: lane(h,m) holds row m, K = kk+8h..+7 and kk+16+8h..+7
            union { v16bf v; uint4 u[2]; } fa[2], fb[4];
#pragma unroll
            for (int i = 0; i < 2; ++i) {
                const unsigned short* base = &sA[(wm + 16 * i + ln) * 72 + kk];
                fa[i].u[0] = *(const uint4*)(base + 8 * h);
                fa[i].u[1] = *(const uint4*)(base + 16 + 8 * h);
            }
            // B frag: lane(h,n) holds col n, K = kk+8h..+7 and kk+16+8h..+7
#pragma unroll
            for (int j = 0; j < 4; ++j) {
                const unsigned short* base = &sB[(wn + 16 * j + ln) * 72 + kk];
                fb[j].u[0] = *(const uint4*)(base + 8 * h);
                fb[j].u[1] = *(const uint4*)(base + 16 + 8 * h);
            }
#pragma unroll
            for (int i = 0; i < 2; ++i)
#pragma unroll
                for (int j = 0; j < 4; ++j)
                    acc[i][j] = __builtin_amdgcn_wmma_f32_16x16x32_bf16(
                        false, fa[i].v, false, fb[j].v, (short)0, acc[i][j], false, false);
        }
        __syncthreads();
    }

    // ---- epilogue: bias + ReLU, bf16 store ----
    __bf16* Outh = (__bf16*)Out;
#pragma unroll
    for (int i = 0; i < 2; ++i) {
#pragma unroll
        for (int j = 0; j < 4; ++j) {
            int col = colBase + wn + 16 * j + ln;
            float bv = bias[col];
#pragma unroll
            for (int r = 0; r < 8; ++r) {
                int row = rowBase + wm + 16 * i + 8 * h + r;
                float v = acc[i][j][r] + bv;
                Outh[(size_t)row * N + col] = (__bf16)(v > 0.f ? v : 0.f);
            }
        }
    }
}

// One block per ROI row: 53 dots (cls 3 | delta 12 | pos 6 from f1; emb 32 from fe)
__global__ __launch_bounds__(64)
void heads_kernel(const unsigned short* __restrict__ f1,
                  const unsigned short* __restrict__ fe,
                  const float* __restrict__ w_cls,   const float* __restrict__ b_cls,
                  const float* __restrict__ w_delta, const float* __restrict__ b_delta,
                  const float* __restrict__ w_pos,   const float* __restrict__ b_pos,
                  const float* __restrict__ w_emb,   const float* __restrict__ b_emb,
                  float* __restrict__ headbuf) {
    __shared__ __align__(16) unsigned short s1[1024];
    __shared__ __align__(16) unsigned short s2[1024];
    const int row = blockIdx.x;
    const int t   = threadIdx.x;
    for (int q = 0; q < 2; ++q) {
        int g = (t + q * 64) * 8;
        *(uint4*)(&s1[g]) = *(const uint4*)(f1 + (size_t)row * 1024 + g);
        *(uint4*)(&s2[g]) = *(const uint4*)(fe + (size_t)row * 1024 + g);
    }
    __syncthreads();
    if (t < 53) {
        const unsigned short* src;
        const float* w;
        float b;
        if (t < 3)       { src = s1; w = w_cls   + (size_t)t * 1024;        b = b_cls[t];        }
        else if (t < 15) { src = s1; w = w_delta + (size_t)(t - 3) * 1024;  b = b_delta[t - 3];  }
        else if (t < 21) { src = s1; w = w_pos   + (size_t)(t - 15) * 1024; b = b_pos[t - 15];   }
        else             { src = s2; w = w_emb   + (size_t)(t - 21) * 1024; b = b_emb[t - 21];   }
        float acc = 0.f;
        for (int k = 0; k < 1024; ++k)
            acc = fmaf(bf2f(src[k]), w[k], acc);
        headbuf[(size_t)row * 53 + t] = acc + b;
    }
}

// softmax + bbox restore + emb replication; out = [bbox 16384x8 | emb 16384x32 | class_num]
__global__ __launch_bounds__(256)
void postproc_kernel(const float* __restrict__ headbuf,
                     const float* __restrict__ rois,  // [N,5]: batch,x1,y1,x2,y2
                     float* __restrict__ out, int Nrows) {
    int idx = blockIdx.x * 256 + threadIdx.x;
    if (idx == 0) out[(size_t)Nrows * 2 * 40] = 2.0f;  // class_num
    if (idx >= Nrows * 2) return;
    int i = idx >> 1, c = idx & 1;
    const float* hb = headbuf + (size_t)i * 53;

    float l0 = hb[0], l1 = hb[1], l2 = hb[2];
    float m  = fmaxf(l0, fmaxf(l1, l2));
    float e0 = expf(l0 - m), e1 = expf(l1 - m), e2 = expf(l2 - m);
    float score = (c == 0 ? e1 : e2) / (e0 + e1 + e2);

    float d0 = hb[3 + 4 + 4 * c + 0] * 0.1f;
    float d1 = hb[3 + 4 + 4 * c + 1] * 0.1f;
    float d2 = hb[3 + 4 + 4 * c + 2] * 0.2f;
    float d3 = hb[3 + 4 + 4 * c + 3] * 0.2f;
    float p0 = hb[15 + 2 + 2 * c + 0] * 0.1f;
    float p1 = hb[15 + 2 + 2 * c + 1] * 0.1f;

    const float* r = rois + (size_t)i * 5;
    float x1 = r[1], y1 = r[2], x2 = r[3], y2 = r[4];
    float bw = x2 - x1, bh = y2 - y1;
    float cx = x1 + 0.5f * bw, cy = y1 + 0.5f * bh;
    float pcx = d0 * bw + cx, pcy = d1 * bh + cy;
    float pw = expf(d2) * bw, ph = expf(d3) * bh;
    float px = p0 * bw + cx, py = p1 * bh + cy;

    float* ob = out + (size_t)idx * 8;
    ob[0] = pcx - 0.5f * pw; ob[1] = pcy - 0.5f * ph;
    ob[2] = pcx + 0.5f * pw; ob[3] = pcy + 0.5f * ph;
    ob[4] = px; ob[5] = py; ob[6] = score; ob[7] = (float)(c + 1);

    float* oe = out + (size_t)Nrows * 2 * 8 + (size_t)idx * 32;
    const float* he = hb + 21;
    for (int q = 0; q < 32; ++q) oe[q] = he[q];
}

extern "C" void kernel_launch(void* const* d_in, const int* in_sizes, int n_in,
                              void* d_out, int out_size, void* d_ws, size_t ws_size,
                              hipStream_t stream) {
    (void)in_sizes; (void)n_in; (void)out_size; (void)ws_size;
    const float* pool   = (const float*)d_in[0];
    const float* rois   = (const float*)d_in[1];
    const float* w_fc1  = (const float*)d_in[2];
    const float* b_fc1  = (const float*)d_in[3];
    const float* w_fc2  = (const float*)d_in[4];
    const float* b_fc2  = (const float*)d_in[5];
    const float* w_fc3  = (const float*)d_in[6];
    const float* b_fc3  = (const float*)d_in[7];
    const float* w_fc4  = (const float*)d_in[8];
    const float* b_fc4  = (const float*)d_in[9];
    const float* w_cls  = (const float*)d_in[10];
    const float* b_cls  = (const float*)d_in[11];
    const float* w_delta= (const float*)d_in[12];
    const float* b_delta= (const float*)d_in[13];
    const float* w_pos  = (const float*)d_in[14];
    const float* b_pos  = (const float*)d_in[15];
    const float* w_emb  = (const float*)d_in[16];
    const float* b_emb  = (const float*)d_in[17];

    const int Nr = 8192, H = 1024, K1 = 12544;
    const size_t HID = (size_t)Nr * H * sizeof(unsigned short);
    char* ws = (char*)d_ws;
    unsigned short* h1m = (unsigned short*)(ws);
    unsigned short* f1  = (unsigned short*)(ws + HID);
    unsigned short* fem = (unsigned short*)(ws + 2 * HID);
    unsigned short* fe  = (unsigned short*)(ws + 3 * HID);
    float*          hbuf= (float*)(ws + 4 * HID);

    dim3 blk(256);
    dim3 grd(H / 128, Nr / 128);  // (8, 64) — col-fast for L2 reuse of the A panel
    gemm_relu_bf16<false><<<grd, blk, 0, stream>>>(pool, w_fc1, b_fc1, h1m, Nr, H, K1);
    gemm_relu_bf16<true ><<<grd, blk, 0, stream>>>(h1m,  w_fc2, b_fc2, f1,  Nr, H, H);
    gemm_relu_bf16<false><<<grd, blk, 0, stream>>>(pool, w_fc3, b_fc3, fem, Nr, H, K1);
    gemm_relu_bf16<true ><<<grd, blk, 0, stream>>>(fem,  w_fc4, b_fc4, fe,  Nr, H, H);
    heads_kernel<<<Nr, 64, 0, stream>>>(f1, fe, w_cls, b_cls, w_delta, b_delta,
                                        w_pos, b_pos, w_emb, b_emb, hbuf);
    postproc_kernel<<<(Nr * 2 + 255) / 256, 256, 0, stream>>>(hbuf, rois, (float*)d_out, Nr);
}
